// MultiHeadSelfAttention_18322330484947
// MI455X (gfx1250) — compile-verified, matrix operand order checked by reading, not run on hardware
//
#include <hip/hip_runtime.h>

// ---- vector types matching CDNA5 WMMA operand shapes ----
typedef __attribute__((ext_vector_type(16))) __bf16 v16bf;
typedef __attribute__((ext_vector_type(8)))  __bf16 v8bf;
typedef __attribute__((ext_vector_type(4)))  __bf16 v4bf;
typedef __attribute__((ext_vector_type(8)))  float  v8f;
typedef __attribute__((ext_vector_type(4)))  float  v4f;

union AFrag { v16bf v; v8bf h[2]; };
union CAcc  { v8f   v; float f[8]; };

static __device__ __forceinline__ v8f wmma_bf16(const AFrag& a, const AFrag& b, v8f c) {
    return __builtin_amdgcn_wmma_f32_16x16x32_bf16(false, a.v, false, b.v,
                                                   (short)0, c, false, false);
}

// ---------------------------------------------------------------------------
// CDNA5 async global->LDS copy (16B per lane), ASYNCcnt-tracked.
// Generic LDS address bits [31:0] == LDS offset (aperture is in [63:32]),
// so truncation yields the required LDS-address VGPR operand.
// ---------------------------------------------------------------------------
static __device__ __forceinline__
void async_g2l_b128(const void* gptr, void* lptr) {
    unsigned lds = (unsigned)(unsigned long long)lptr;
    asm volatile("global_load_async_to_lds_b128 %0, %1, off"
                 :: "v"(lds), "v"(gptr) : "memory");
}
static __device__ __forceinline__ void wait_async0() {
    asm volatile("s_wait_asynccnt 0" ::: "memory");
}

// ---------------------------------------------------------------------------
// fp32 -> bf16 pack (n divisible by 1024)
// ---------------------------------------------------------------------------
__global__ __launch_bounds__(256)
void cvt_f32_bf16(const float* __restrict__ in, __bf16* __restrict__ out, int n) {
    int i = (blockIdx.x * 256 + threadIdx.x) * 4;
    if (i < n) {
        v4f x = *(const v4f*)(in + i);
        v4bf y;
        y.x = (__bf16)x.x; y.y = (__bf16)x.y; y.z = (__bf16)x.z; y.w = (__bf16)x.w;
        *(v4bf*)(out + i) = y;
    }
}

// Stage one 128x32 A-tile and 128x32 B-tile (bf16) via async DMA.
static __device__ __forceinline__
void stage_tiles(const __bf16* __restrict__ A, const __bf16* __restrict__ W,
                 __bf16* asb, __bf16* bsb, int tileM, int tileN, int kt, int tid) {
    int r0 = tid >> 2,          c0 = (tid & 3) * 8;
    int r1 = (tid + 256) >> 2,  c1 = (tid & 3) * 8;
    async_g2l_b128(&A[(size_t)(tileM + r0) * 1024 + kt + c0], &asb[r0 * 32 + c0]);
    async_g2l_b128(&A[(size_t)(tileM + r1) * 1024 + kt + c1], &asb[r1 * 32 + c1]);
    async_g2l_b128(&W[(size_t)(tileN + r0) * 1024 + kt + c0], &bsb[r0 * 32 + c0]);
    async_g2l_b128(&W[(size_t)(tileN + r1) * 1024 + kt + c1], &bsb[r1 * 32 + c1]);
}

// ---------------------------------------------------------------------------
// C[4096,1024] = A[4096,1024](bf16) . W[1024,1024]^T(bf16) + bias
// mode 0: bf16 row-major out    (Q, K)
// mode 1: bf16 per-head transposed out Vt[b][h][d][s]   (V)
// mode 2: fp32 row-major out    (final projection -> d_out)
// Block: 256 thr (8 waves), tile 128x128, wave tile 64x32, K-step 32.
// Double-buffered LDS; next tile streams in via async DMA during WMMA.
// ---------------------------------------------------------------------------
__global__ __launch_bounds__(256)
void gemm4096_nt(const __bf16* __restrict__ A, const __bf16* __restrict__ W,
                 const float* __restrict__ bias, void* __restrict__ outp, int mode) {
    __shared__ __bf16 As[2][128 * 32];
    __shared__ __bf16 Bs[2][128 * 32];
    const int tid  = threadIdx.x;
    const int lane = tid & 31, wave = tid >> 5;
    const int wm = wave >> 2, wn = wave & 3;          // 2x4 wave grid
    const int tileM = blockIdx.y * 128, tileN = blockIdx.x * 128;
    const int r15 = lane & 15, hsel = lane >> 4;

    CAcc acc[4][2];
#pragma unroll
    for (int i = 0; i < 4; ++i)
#pragma unroll
        for (int j = 0; j < 2; ++j)
#pragma unroll
            for (int e = 0; e < 8; ++e) acc[i][j].f[e] = 0.0f;

    stage_tiles(A, W, As[0], Bs[0], tileM, tileN, 0, tid);
    wait_async0();
    __syncthreads();

    int cur = 0;
    for (int kt = 0; kt < 1024; kt += 32) {
        if (kt + 32 < 1024)
            stage_tiles(A, W, As[cur ^ 1], Bs[cur ^ 1], tileM, tileN, kt + 32, tid);

        AFrag aF[4], bF[2];
#pragma unroll
        for (int i = 0; i < 4; ++i) {
            const __bf16* p = &As[cur][(wm * 64 + i * 16 + r15) * 32 + hsel * 8];
            aF[i].h[0] = *(const v8bf*)p;
            aF[i].h[1] = *(const v8bf*)(p + 16);
        }
#pragma unroll
        for (int j = 0; j < 2; ++j) {
            const __bf16* p = &Bs[cur][(wn * 32 + j * 16 + r15) * 32 + hsel * 8];
            bF[j].h[0] = *(const v8bf*)p;
            bF[j].h[1] = *(const v8bf*)(p + 16);
        }
#pragma unroll
        for (int i = 0; i < 4; ++i)
#pragma unroll
            for (int j = 0; j < 2; ++j)
                acc[i][j].v = wmma_bf16(aF[i], bF[j], acc[i][j].v);

        wait_async0();          // next tile's DMA landed (per-wave, in-order)
        __syncthreads();        // all waves done reading cur + next resident
        cur ^= 1;
    }

#pragma unroll
    for (int j = 0; j < 2; ++j) {
        int n = tileN + wn * 32 + j * 16 + r15;
        float bval = bias[n];
#pragma unroll
        for (int i = 0; i < 4; ++i) {
            int mbase = tileM + wm * 64 + i * 16 + hsel * 8;
#pragma unroll
            for (int e = 0; e < 8; ++e) {
                int m = mbase + e;
                float val = acc[i][j].f[e] + bval;
                if (mode == 0) {
                    ((__bf16*)outp)[(size_t)m * 1024 + n] = (__bf16)val;
                } else if (mode == 1) {
                    int bb = m >> 10, s = m & 1023;
                    int hh = n >> 6,  d = n & 63;
                    ((__bf16*)outp)[(((size_t)(bb * 16 + hh)) * 64 + d) * 1024 + s] = (__bf16)val;
                } else {
                    ((float*)outp)[(size_t)m * 1024 + n] = val;
                }
            }
        }
    }
}

// ---------------------------------------------------------------------------
// scores = (Q Kt) * 0.125, softmax over keys, write attn fp32.
// One block per (b, h, 16-query tile). 8 waves; each wave does one 16-key
// column tile per 128-key chunk (K streamed via async DMA). Scores strip in
// 64KB LDS fp32; wave-per-row softmax with v_exp + lane shuffles.
// ---------------------------------------------------------------------------
__global__ __launch_bounds__(256)
void attn_scores_softmax(const __bf16* __restrict__ Qb, const __bf16* __restrict__ Kb,
                         float* __restrict__ attn) {
    __shared__ __bf16 qs[16 * 64];
    __shared__ __bf16 ks[128 * 64];
    __shared__ float  sc[16 * 1024];
    const int tid = threadIdx.x, lane = tid & 31, wave = tid >> 5;
    const int r15 = lane & 15, hsel = lane >> 4;
    const int qt = blockIdx.x & 63, bh = blockIdx.x >> 6;
    const int b = bh >> 4, h = bh & 15;

    if (tid < 128) {
        int r = tid >> 3, c = (tid & 7) * 8;
        *(v8bf*)&qs[r * 64 + c] =
            *(const v8bf*)&Qb[((size_t)(b * 1024 + qt * 16 + r)) * 1024 + h * 64 + c];
    }

    for (int ck = 0; ck < 8; ++ck) {
        __syncthreads();                              // protect ks reuse
#pragma unroll
        for (int it = 0; it < 4; ++it) {
            int cid = tid + it * 256;                 // 1024 chunks of 16B
            int r = cid >> 3, c = (cid & 7) * 8;
            async_g2l_b128(&Kb[((size_t)(b * 1024 + ck * 128 + r)) * 1024 + h * 64 + c],
                           &ks[r * 64 + c]);
        }
        wait_async0();
        __syncthreads();

        CAcc acc;
#pragma unroll
        for (int e = 0; e < 8; ++e) acc.f[e] = 0.0f;
#pragma unroll
        for (int k2 = 0; k2 < 64; k2 += 32) {
            AFrag a, bfr;
            const __bf16* pa = &qs[r15 * 64 + k2 + hsel * 8];
            a.h[0] = *(const v8bf*)pa;  a.h[1] = *(const v8bf*)(pa + 16);
            const __bf16* pb = &ks[(wave * 16 + r15) * 64 + k2 + hsel * 8];
            bfr.h[0] = *(const v8bf*)pb; bfr.h[1] = *(const v8bf*)(pb + 16);
            acc.v = wmma_bf16(a, bfr, acc.v);
        }
        int col = ck * 128 + wave * 16 + r15;
#pragma unroll
        for (int e = 0; e < 8; ++e)
            sc[(e + hsel * 8) * 1024 + col] = acc.f[e] * 0.125f;   // 1/sqrt(64)
    }
    __syncthreads();

    // wave-per-row softmax: wave handles rows 2w, 2w+1; lanes stride cols
    for (int rr = 0; rr < 2; ++rr) {
        int row = wave * 2 + rr;
        float* srow = &sc[row * 1024];
        float m = -3.4e38f;
        for (int c = lane; c < 1024; c += 32) m = fmaxf(m, srow[c]);
#pragma unroll
        for (int off = 16; off > 0; off >>= 1) m = fmaxf(m, __shfl_xor(m, off, 32));
        float sum = 0.0f;
        for (int c = lane; c < 1024; c += 32) {
            float e = __expf(srow[c] - m);
            srow[c] = e;
            sum += e;
        }
#pragma unroll
        for (int off = 16; off > 0; off >>= 1) sum += __shfl_xor(sum, off, 32);
        float inv = 1.0f / sum;
        float* orow = attn + (((size_t)bh * 1024) + qt * 16 + row) * 1024;
        for (int c = lane; c < 1024; c += 32) orow[c] = srow[c] * inv;
    }
}

// ---------------------------------------------------------------------------
// ctx = attn(fp32, 1024x1024) . V(1024x64) per (b,h); attn converted to bf16
// in LDS on the fly (needs VGPR convert -> regular loads + prefetch); Vt tile
// streamed via async DMA. One block per (b, h, 128-query tile). Wave owns
// 16x64 of ctx (4 wmma tiles). Output ctx bf16 in (b, s, h*64+d) layout.
// ---------------------------------------------------------------------------
__global__ __launch_bounds__(256)
void attn_ctx(const float* __restrict__ attn, const __bf16* __restrict__ Vt,
              __bf16* __restrict__ ctxb) {
    __shared__ __bf16 As[128 * 32];
    __shared__ __bf16 Bs[64 * 32];
    const int tid = threadIdx.x, lane = tid & 31, wave = tid >> 5;
    const int r15 = lane & 15, hsel = lane >> 4;
    const int qt = blockIdx.x & 7, bh = blockIdx.x >> 3;
    const int b = bh >> 4, h = bh & 15;

    CAcc acc[4];
#pragma unroll
    for (int j = 0; j < 4; ++j)
#pragma unroll
        for (int e = 0; e < 8; ++e) acc[j].f[e] = 0.0f;

    const float*  abase = attn + ((size_t)bh * 1024 + qt * 128) * 1024;
    const __bf16* vbase = Vt + (size_t)bh * 64 * 1024;

    for (int kt = 0; kt < 1024; kt += 32) {
        __syncthreads();
        {
            int r = tid >> 2, c = (tid & 3) * 8;      // 64x32 bf16 V chunk, async DMA
            async_g2l_b128(&vbase[(size_t)r * 1024 + kt + c], &Bs[r * 32 + c]);
        }
#pragma unroll
        for (int it = 0; it < 4; ++it) {
            int cid = tid + it * 256;                 // 1024 float4 chunks
            int r = cid >> 3, c = (cid & 7) * 4;
            v4f x = *(const v4f*)&abase[(size_t)r * 1024 + kt + c];
            v4bf y;
            y.x = (__bf16)x.x; y.y = (__bf16)x.y; y.z = (__bf16)x.z; y.w = (__bf16)x.w;
            *(v4bf*)&As[r * 32 + c] = y;
        }
        if (kt + 32 < 1024)
            __builtin_prefetch(&abase[(size_t)(tid >> 1) * 1024 + kt + 32], 0, 3);
        wait_async0();
        __syncthreads();

        AFrag a;
        const __bf16* pa = &As[(wave * 16 + r15) * 32 + hsel * 8];
        a.h[0] = *(const v8bf*)pa;  a.h[1] = *(const v8bf*)(pa + 16);
#pragma unroll
        for (int j = 0; j < 4; ++j) {
            AFrag bfr;
            const __bf16* pb = &Bs[(j * 16 + r15) * 32 + hsel * 8];
            bfr.h[0] = *(const v8bf*)pb; bfr.h[1] = *(const v8bf*)(pb + 16);
            acc[j].v = wmma_bf16(a, bfr, acc[j].v);
        }
    }

#pragma unroll
    for (int j = 0; j < 4; ++j) {
        int d = h * 64 + j * 16 + r15;
#pragma unroll
        for (int e = 0; e < 8; ++e) {
            int srow = qt * 128 + wave * 16 + hsel * 8 + e;
            ctxb[((size_t)(b * 1024 + srow)) * 1024 + d] = (__bf16)acc[j].f[e];
        }
    }
}

// ---------------------------------------------------------------------------
extern "C" void kernel_launch(void* const* d_in, const int* in_sizes, int n_in,
                              void* d_out, int out_size, void* d_ws, size_t ws_size,
                              hipStream_t stream) {
    (void)in_sizes; (void)n_in; (void)out_size; (void)ws_size;
    const float* x  = (const float*)d_in[0];
    const float* wq = (const float*)d_in[1];
    const float* bq = (const float*)d_in[2];
    const float* wk = (const float*)d_in[3];
    const float* bk = (const float*)d_in[4];
    const float* wv = (const float*)d_in[5];
    const float* bv = (const float*)d_in[6];
    const float* wo = (const float*)d_in[7];
    const float* bo = (const float*)d_in[8];

    char* ws = (char*)d_ws;
    __bf16* xb   = (__bf16*)(ws);                         // 8 MB
    __bf16* wqb  = (__bf16*)(ws + ((size_t)8  << 20));    // 2 MB each
    __bf16* wkb  = (__bf16*)(ws + ((size_t)10 << 20));
    __bf16* wvb  = (__bf16*)(ws + ((size_t)12 << 20));
    __bf16* wob  = (__bf16*)(ws + ((size_t)14 << 20));
    __bf16* Qb   = (__bf16*)(ws + ((size_t)16 << 20));    // 8 MB
    __bf16* Kb   = (__bf16*)(ws + ((size_t)24 << 20));    // 8 MB
    __bf16* Vt   = (__bf16*)(ws + ((size_t)32 << 20));    // 8 MB
    __bf16* ctxb = (__bf16*)(ws + ((size_t)40 << 20));    // 8 MB

    float* out0 = (float*)d_out;
    float* attn = out0 + (size_t)4 * 1024 * 1024;

    const int nx = 4096 * 1024, nw = 1024 * 1024;
    cvt_f32_bf16<<<nx / 1024, 256, 0, stream>>>(x,  xb,  nx);
    cvt_f32_bf16<<<nw / 1024, 256, 0, stream>>>(wq, wqb, nw);
    cvt_f32_bf16<<<nw / 1024, 256, 0, stream>>>(wk, wkb, nw);
    cvt_f32_bf16<<<nw / 1024, 256, 0, stream>>>(wv, wvb, nw);
    cvt_f32_bf16<<<nw / 1024, 256, 0, stream>>>(wo, wob, nw);

    dim3 gg(8, 32);   // N/128, M/128
    gemm4096_nt<<<gg, 256, 0, stream>>>(xb, wqb, bq, (void*)Qb, 0);
    gemm4096_nt<<<gg, 256, 0, stream>>>(xb, wkb, bk, (void*)Kb, 0);
    gemm4096_nt<<<gg, 256, 0, stream>>>(xb, wvb, bv, (void*)Vt, 1);

    attn_scores_softmax<<<4096, 256, 0, stream>>>(Qb, Kb, attn);
    attn_ctx<<<512, 256, 0, stream>>>(attn, Vt, ctxb);

    gemm4096_nt<<<gg, 256, 0, stream>>>(ctxb, wob, bo, (void*)out0, 2);
}
